// GPT_49323404427743
// MI455X (gfx1250) — compile-verified, matrix operand order checked by reading, not run on hardware
//
#include <hip/hip_runtime.h>
#include <hip/hip_bf16.h>
#include <stdint.h>

// ---------------------------------------------------------------- types ----
typedef __attribute__((ext_vector_type(16))) __bf16 v16bf;
typedef __attribute__((ext_vector_type(8)))  float  v8f;

#define LAYERS 6
#define DMODEL 768
#define NHEAD  12
#define DHEAD  64
#define FFDIM  3072
#define VOCAB  32000
#define SEQ    1024
#define MTOK   2048            // B*T

union FragU {
    uint32_t u[8];
    uint4    q4[2];
    v16bf    bf;
};

__device__ __forceinline__ v8f v8f_zero() {
    v8f z;
#pragma unroll
    for (int i = 0; i < 8; ++i) z[i] = 0.f;
    return z;
}

__device__ __forceinline__ uint32_t pk_bf16(float lo, float hi) {
    union { float f; uint32_t u; } a, b;
    a.f = lo; b.f = hi;
    uint32_t x = (a.u + 0x7FFFu + ((a.u >> 16) & 1u)) >> 16;   // RNE
    uint32_t y = (b.u + 0x7FFFu + ((b.u >> 16) & 1u)) >> 16;
    return (x & 0xFFFFu) | (y << 16);
}

__device__ __forceinline__ uint16_t f32_to_bf16(float f) {
    union { float f; uint32_t u; } a; a.f = f;
    return (uint16_t)((a.u + 0x7FFFu + ((a.u >> 16) & 1u)) >> 16);
}

// Load one 16x32 bf16 A/B fragment.  p must already point at
// (row base) + k0 + 8*halfLane, 16B aligned.  ISA layout (CDNA5 ISA):
//  lanes 0-15: VGPR0-3 = K[k0..k0+7], VGPR4-7 = K[k0+16..k0+23]
//  lanes 16-31: same +8.
__device__ __forceinline__ void load_frag16(FragU& f, const uint16_t* p) {
    f.q4[0] = *(const uint4*)p;
    f.q4[1] = *(const uint4*)(p + 16);
}

__device__ __forceinline__ v8f wmma_bf16(const FragU& a, const FragU& b, v8f c) {
    return __builtin_amdgcn_wmma_f32_16x16x32_bf16(
        /*neg_a=*/false, a.bf, /*neg_b=*/false, b.bf,
        /*c_mod=*/(short)0, c, /*reuse_a=*/false, /*reuse_b=*/false);
}

// CDNA5 async global->LDS copy (ASYNCcnt path, bypasses VGPRs).
// IOFFSET is added to BOTH the LDS dest (VGPR[VDST]) and the global address
// (VGPR[VADDR]) per CDNA5 ISA, so one base pair covers
// a whole 128B row with immediate offsets.  lds = low 32 bits of the generic
// LDS address (== LDS byte offset per the aperture rules).
__device__ __forceinline__ void async_copy_row128(uint32_t lds, const uint16_t* g) {
    uint64_t ga = (uint64_t)(uintptr_t)g;
    asm volatile(
        "global_load_async_to_lds_b128 %0, %1, off\n\t"
        "global_load_async_to_lds_b128 %0, %1, off offset:16\n\t"
        "global_load_async_to_lds_b128 %0, %1, off offset:32\n\t"
        "global_load_async_to_lds_b128 %0, %1, off offset:48\n\t"
        "global_load_async_to_lds_b128 %0, %1, off offset:64\n\t"
        "global_load_async_to_lds_b128 %0, %1, off offset:80\n\t"
        "global_load_async_to_lds_b128 %0, %1, off offset:96\n\t"
        "global_load_async_to_lds_b128 %0, %1, off offset:112"
        :: "v"(lds), "v"(ga) : "memory");
}

// ------------------------------------------------------------- embedding ---
__global__ __launch_bounds__(256) void embed_kernel(
    const int* __restrict__ X, const float* __restrict__ Tok,
    const float* __restrict__ Pos, float* __restrict__ Hout)
{
    const int row = blockIdx.x;
    const int t   = row & (SEQ - 1);
    const int id  = X[row];
    for (int i = threadIdx.x; i < DMODEL; i += 256)
        Hout[(size_t)row * DMODEL + i] =
            Tok[(size_t)id * DMODEL + i] + Pos[(size_t)t * DMODEL + i];
}

// ------------------------------------------------------------- layernorm ---
__global__ __launch_bounds__(256) void ln_kernel(
    const float* __restrict__ X, const float* __restrict__ G,
    const float* __restrict__ Bv, uint16_t* __restrict__ Out, int D)
{
    __shared__ float red[256];
    const int row = blockIdx.x, tid = threadIdx.x;
    const float* xr = X + (size_t)row * D;

    float s = 0.f;
    for (int i = tid; i < D; i += 256) s += xr[i];
    red[tid] = s; __syncthreads();
    for (int o = 128; o > 0; o >>= 1) { if (tid < o) red[tid] += red[tid + o]; __syncthreads(); }
    const float mean = red[0] / (float)D;
    __syncthreads();

    float v = 0.f;
    for (int i = tid; i < D; i += 256) { float d = xr[i] - mean; v += d * d; }
    red[tid] = v; __syncthreads();
    for (int o = 128; o > 0; o >>= 1) { if (tid < o) red[tid] += red[tid + o]; __syncthreads(); }
    const float rstd = rsqrtf(red[0] / (float)D + 1e-5f);

    for (int i = tid; i < D; i += 256)
        Out[(size_t)row * D + i] = f32_to_bf16((xr[i] - mean) * rstd * G[i] + Bv[i]);
}

// ---------------------------------------------------------------- GEMM -----
// out[M,N] = A[M,K](bf16) @ W[N,K](fp32 -> bf16 in-kernel) + bias
// A tile is staged by async global->LDS DMA, double-buffered against compute.
// mode 0: fp32 out (+optional residual, may alias out)      [wo, w2, head]
// mode 1: bf16 out (+optional ReLU)                          [q, k, fc1]
// mode 2: bf16 transposed scatter -> V^T [B,H,DK,T]          [v]
#define GBM 256
#define GBN 64
#define GBK 64
#define GPAD 8   // 144B LDS row stride: 16-lane b128 reads conflict-free

__global__ __launch_bounds__(256) void gemm_nt(
    const uint16_t* __restrict__ A, const float* __restrict__ W,
    const float* __restrict__ bias, const float* __restrict__ resid,
    float* __restrict__ outF, uint16_t* __restrict__ outH,
    int N, int K, int mode, int relu)
{
    __shared__ __align__(16) uint16_t As[2][GBM][GBK + GPAD];  // double buffer
    __shared__ __align__(16) uint16_t Bs[GBN][GBK + GPAD];

    const int tid  = threadIdx.x;
    const int lane = tid & 31, wid = tid >> 5;
    const int hf   = lane >> 4, l16 = lane & 15;
    const int m0   = blockIdx.y * GBM;
    const int n0   = blockIdx.x * GBN;

    v8f acc[2][4];
#pragma unroll
    for (int i = 0; i < 2; ++i)
#pragma unroll
        for (int j = 0; j < 4; ++j) acc[i][j] = v8f_zero();

    const int brow = tid >> 2, bseg = tid & 3;

    // per-thread async staging state: thread owns A row m0+tid
    const uint16_t* arow = A + (size_t)(m0 + tid) * K;
    const uint32_t ldsA[2] = { (uint32_t)(uintptr_t)&As[0][tid][0],
                               (uint32_t)(uintptr_t)&As[1][tid][0] };

    const int nchunks = K / GBK;
    async_copy_row128(ldsA[0], arow);              // prefetch chunk 0

    for (int c = 0; c < nchunks; ++c) {
        const int p  = c & 1;
        const int k0 = c * GBK;

        // stage B tile with fp32 -> bf16 conversion (bandwidth-optimal:
        // weights are fp32 in HBM, read once at 4B/elt, never re-written)
        {
            const float* src = W + (size_t)(n0 + brow) * K + k0 + bseg * 16;
            uint32_t d[8];
#pragma unroll
            for (int i = 0; i < 4; ++i) {
                float4 f = ((const float4*)src)[i];
                d[2 * i + 0] = pk_bf16(f.x, f.y);
                d[2 * i + 1] = pk_bf16(f.z, f.w);
            }
            uint4* dst = (uint4*)&Bs[brow][bseg * 16];
            dst[0] = make_uint4(d[0], d[1], d[2], d[3]);
            dst[1] = make_uint4(d[4], d[5], d[6], d[7]);
        }

        // kick off next A chunk into the other buffer, then wait for the
        // current chunk (async loads complete in order; 8 in flight = next).
        if (c + 1 < nchunks) {
            async_copy_row128(ldsA[p ^ 1], arow + (size_t)(c + 1) * GBK);
            asm volatile("s_wait_asynccnt 0x8" ::: "memory");
        } else {
            asm volatile("s_wait_asynccnt 0x0" ::: "memory");
        }
        __syncthreads();   // all waves' async A data + Bs visible

#pragma unroll
        for (int kk = 0; kk < GBK; kk += 32) {
            FragU af[2], bf[4];
#pragma unroll
            for (int mt = 0; mt < 2; ++mt)
                load_frag16(af[mt], &As[p][wid * 32 + mt * 16 + l16][kk + 8 * hf]);
#pragma unroll
            for (int nt = 0; nt < 4; ++nt)
                load_frag16(bf[nt], &Bs[nt * 16 + l16][kk + 8 * hf]);
#pragma unroll
            for (int mt = 0; mt < 2; ++mt)
#pragma unroll
                for (int nt = 0; nt < 4; ++nt)
                    acc[mt][nt] = wmma_bf16(af[mt], bf[nt], acc[mt][nt]);
        }
        __syncthreads();   // Bs reuse / buffer rotation safety
    }

    // epilogue (C/D layout: VGPR r, lanes0-15 -> M=r,N=lane; lanes16-31 -> M=8+r)
#pragma unroll
    for (int nt = 0; nt < 4; ++nt) {
        const int col = n0 + nt * 16 + l16;
        const float bv = bias ? bias[col] : 0.f;
#pragma unroll
        for (int mt = 0; mt < 2; ++mt) {
#pragma unroll
            for (int r = 0; r < 8; ++r) {
                const int row = m0 + wid * 32 + mt * 16 + 8 * hf + r;
                float v = acc[mt][nt][r] + bv;
                if (mode == 0) {
                    if (resid) v += resid[(size_t)row * N + col];
                    outF[(size_t)row * N + col] = v;
                } else if (mode == 1) {
                    if (relu) v = fmaxf(v, 0.f);
                    outH[(size_t)row * N + col] = f32_to_bf16(v);
                } else {  // mode 2: V^T scatter [B,H,DK,T]
                    const int b  = row >> 10, t = row & 1023;
                    const int hh = col >> 6,  dk = col & 63;
                    outH[(((size_t)(b * NHEAD + hh)) * DHEAD + dk) * SEQ + t] =
                        f32_to_bf16(v);
                }
            }
        }
    }
}

// ------------------------------------------------------------- attention ---
// Flash-style causal attention. One wave per 16-row query tile.
// Q,K: bf16 [B,T,D] (head slice = contiguous 64); V^T: bf16 [B,H,DK,T].
__global__ __launch_bounds__(128) void attn_kernel(
    const uint16_t* __restrict__ Q, const uint16_t* __restrict__ Kb,
    const uint16_t* __restrict__ Vt, uint16_t* __restrict__ Ctx)
{
    __shared__ __align__(16) float Pst[4][16][40];   // per-wave prob staging
    const int lane = threadIdx.x & 31, widx = threadIdx.x >> 5;
    const int hf = lane >> 4, l16 = lane & 15;
    const int qt = blockIdx.x / 6;                 // all waves in block share qt
    const int bh = (blockIdx.x % 6) * 4 + widx;    // 24 (b,h) pairs
    const int b = bh / NHEAD, h = bh % NHEAD;
    const int q0 = qt * 16;
    const float scale = 0.125f;                    // 1/sqrt(DK)

    FragU qf0, qf1;
    {
        const size_t rowQ = (size_t)(b * SEQ + q0 + l16) * DMODEL + h * DHEAD;
        load_frag16(qf0, Q + rowQ + 0  + 8 * hf);
        load_frag16(qf1, Q + rowQ + 32 + 8 * hf);
    }

    v8f ctxa[4];
    float mrow[8], lrow[8];
#pragma unroll
    for (int nt = 0; nt < 4; ++nt) ctxa[nt] = v8f_zero();
#pragma unroll
    for (int r = 0; r < 8; ++r) { mrow[r] = -1e30f; lrow[r] = 0.f; }

    const int nkt = (q0 + 16 + 31) >> 5;           // key tiles of 32
    for (int kt = 0; kt < nkt; ++kt) {
        const int kb = kt * 32;

        // scores 16x32 = two N=16 WMMAs, each chaining DK=64 as 2 K-steps
        v8f s[2];
#pragma unroll
        for (int j = 0; j < 2; ++j) {
            FragU kf0, kf1;
            const size_t rowK =
                (size_t)(b * SEQ + kb + j * 16 + l16) * DMODEL + h * DHEAD;
            load_frag16(kf0, Kb + rowK + 0  + 8 * hf);
            load_frag16(kf1, Kb + rowK + 32 + 8 * hf);
            s[j] = wmma_bf16(qf1, kf1, wmma_bf16(qf0, kf0, v8f_zero()));
        }

        // causal mask + online softmax (row groups live in 16-lane halves)
#pragma unroll
        for (int r = 0; r < 8; ++r) {
            const int row = q0 + 8 * hf + r;
            float s0 = s[0][r] * scale; if (kb + l16 > row)      s0 = -1e30f;
            float s1 = s[1][r] * scale; if (kb + 16 + l16 > row) s1 = -1e30f;
            float mx = fmaxf(s0, s1);
#pragma unroll
            for (int off = 1; off < 16; off <<= 1)
                mx = fmaxf(mx, __shfl_xor(mx, off, 32));
            const float mnew = fmaxf(mrow[r], mx);
            const float corr = __expf(mrow[r] - mnew);
            const float p0 = __expf(s0 - mnew);
            const float p1 = __expf(s1 - mnew);
            float ps = p0 + p1;
#pragma unroll
            for (int off = 1; off < 16; off <<= 1)
                ps += __shfl_xor(ps, off, 32);
            lrow[r] = lrow[r] * corr + ps;
            mrow[r] = mnew;
#pragma unroll
            for (int nt = 0; nt < 4; ++nt) ctxa[nt][r] *= corr;
            Pst[widx][8 * hf + r][l16]      = p0;
            Pst[widx][8 * hf + r][16 + l16] = p1;
        }
        __builtin_amdgcn_wave_barrier();   // LDS ops are in-order per wave;
                                           // this stops compiler reordering
        // reload probs in A-fragment layout, converting to bf16
        FragU pf;
        {
            const float* pr = &Pst[widx][l16][8 * hf];
            float4 f0 = *(const float4*)(pr + 0);
            float4 f1 = *(const float4*)(pr + 4);
            float4 f2 = *(const float4*)(pr + 16);
            float4 f3 = *(const float4*)(pr + 20);
            pf.u[0] = pk_bf16(f0.x, f0.y); pf.u[1] = pk_bf16(f0.z, f0.w);
            pf.u[2] = pk_bf16(f1.x, f1.y); pf.u[3] = pk_bf16(f1.z, f1.w);
            pf.u[4] = pk_bf16(f2.x, f2.y); pf.u[5] = pk_bf16(f2.z, f2.w);
            pf.u[6] = pk_bf16(f3.x, f3.y); pf.u[7] = pk_bf16(f3.z, f3.w);
        }
        __builtin_amdgcn_wave_barrier();

        // ctx += P(16x32keys) @ V(32keys x 64dk): B-frags contiguous from V^T
#pragma unroll
        for (int nt = 0; nt < 4; ++nt) {
            FragU vf;
            const size_t rowV =
                (size_t)((b * NHEAD + h) * DHEAD + nt * 16 + l16) * SEQ + kb;
            load_frag16(vf, Vt + rowV + 8 * hf);
            ctxa[nt] = wmma_bf16(pf, vf, ctxa[nt]);
        }
    }

    // normalize + store bf16 ctx [B,T,D]
#pragma unroll
    for (int nt = 0; nt < 4; ++nt)
#pragma unroll
        for (int r = 0; r < 8; ++r) {
            const int row = q0 + 8 * hf + r;
            const float v = ctxa[nt][r] / lrow[r];
            Ctx[(size_t)(b * SEQ + row) * DMODEL + h * DHEAD + nt * 16 + l16] =
                f32_to_bf16(v);
        }
}

// ---------------------------------------------------------------- driver ---
extern "C" void kernel_launch(void* const* d_in, const int* in_sizes, int n_in,
                              void* d_out, int out_size, void* d_ws, size_t ws_size,
                              hipStream_t stream)
{
    (void)in_sizes; (void)n_in; (void)out_size; (void)ws_size;

    const int*   x       = (const int*)  d_in[0];
    const float* tok_emb = (const float*)d_in[1];
    const float* pos_emb = (const float*)d_in[2];
    const float* ln1_w   = (const float*)d_in[3];
    const float* ln1_b   = (const float*)d_in[4];
    const float* wq = (const float*)d_in[5];  const float* bq = (const float*)d_in[6];
    const float* wk = (const float*)d_in[7];  const float* bk = (const float*)d_in[8];
    const float* wv = (const float*)d_in[9];  const float* bv = (const float*)d_in[10];
    const float* wo = (const float*)d_in[11]; const float* bo = (const float*)d_in[12];
    const float* ln2_w = (const float*)d_in[13]; const float* ln2_b = (const float*)d_in[14];
    const float* w1 = (const float*)d_in[15]; const float* b1 = (const float*)d_in[16];
    const float* w2 = (const float*)d_in[17]; const float* b2 = (const float*)d_in[18];
    const float* lnf_w = (const float*)d_in[19]; const float* lnf_b = (const float*)d_in[20];
    const float* head_w = (const float*)d_in[21];

    char* ws = (char*)d_ws;
    float*    h   = (float*)   (ws + 0);          // 2048*768*4  = 6291456
    uint16_t* a   = (uint16_t*)(ws + 6291456);    // 2048*768*2  = 3145728
    uint16_t* qb  = (uint16_t*)(ws + 9437184);
    uint16_t* kb  = (uint16_t*)(ws + 12582912);
    uint16_t* vt  = (uint16_t*)(ws + 15728640);
    uint16_t* ctx = (uint16_t*)(ws + 18874368);
    uint16_t* ff  = (uint16_t*)(ws + 22020096);   // 2048*3072*2 = 12582912
    float* out = (float*)d_out;

    dim3 gD(DMODEL / GBN, MTOK / GBM);  // (12, 8)
    dim3 gF(FFDIM  / GBN, MTOK / GBM);  // (48, 8)
    dim3 gV(VOCAB  / GBN, MTOK / GBM);  // (500, 8)

    embed_kernel<<<MTOK, 256, 0, stream>>>(x, tok_emb, pos_emb, h);

    for (int i = 0; i < LAYERS; ++i) {
        const size_t wD = (size_t)i * DMODEL * DMODEL;
        ln_kernel<<<MTOK, 256, 0, stream>>>(h, ln1_w + i * DMODEL, ln1_b + i * DMODEL, a, DMODEL);
        gemm_nt<<<gD, 256, 0, stream>>>(a, wq + wD, bq + i * DMODEL, nullptr,
                                        nullptr, qb, DMODEL, DMODEL, 1, 0);
        gemm_nt<<<gD, 256, 0, stream>>>(a, wk + wD, bk + i * DMODEL, nullptr,
                                        nullptr, kb, DMODEL, DMODEL, 1, 0);
        gemm_nt<<<gD, 256, 0, stream>>>(a, wv + wD, bv + i * DMODEL, nullptr,
                                        nullptr, vt, DMODEL, DMODEL, 2, 0);
        attn_kernel<<<384, 128, 0, stream>>>(qb, kb, vt, ctx);
        gemm_nt<<<gD, 256, 0, stream>>>(ctx, wo + wD, bo + i * DMODEL, h,
                                        h, nullptr, DMODEL, DMODEL, 0, 0);
        ln_kernel<<<MTOK, 256, 0, stream>>>(h, ln2_w + i * DMODEL, ln2_b + i * DMODEL, a, DMODEL);
        gemm_nt<<<gF, 256, 0, stream>>>(a, w1 + (size_t)i * FFDIM * DMODEL,
                                        b1 + i * FFDIM, nullptr,
                                        nullptr, ff, FFDIM, DMODEL, 1, 1);
        gemm_nt<<<gD, 256, 0, stream>>>(ff, w2 + (size_t)i * DMODEL * FFDIM,
                                        b2 + i * DMODEL, h,
                                        h, nullptr, DMODEL, FFDIM, 0, 0);
    }

    ln_kernel<<<MTOK, 256, 0, stream>>>(h, lnf_w, lnf_b, a, DMODEL);
    gemm_nt<<<gV, 256, 0, stream>>>(a, head_w, nullptr, nullptr,
                                    out, nullptr, VOCAB, DMODEL, 0, 0);
}
